// GemNetEncoder_79714593014005
// MI455X (gfx1250) — matrix operand order repaired and checked by reading, not compiled, not run on hardware
//
#include <hip/hip_runtime.h>
#include <hip/hip_bf16.h>
#include <math.h>

typedef __attribute__((ext_vector_type(16))) _Float16 v16h;
typedef __attribute__((ext_vector_type(8)))  _Float16 v8h;
typedef __attribute__((ext_vector_type(8)))  float    v8f;

#define HID 128
#define EDGE 256
#define RAD 64
#define CUTOFF_F 6.0f
#define MTILE 32   // edges per block in msg_kernel (2 x 16-row WMMA M-tiles)

// ---------------- WMMA helpers ----------------

__device__ inline v8f wmma16(v16h a, v16h b, v8f c) {
  // emits v_wmma_f32_16x16x32_f16
  return __builtin_amdgcn_wmma_f32_16x16x32_f16(
      /*neg_a=*/false, a, /*neg_b=*/false, b,
      /*c_mod=*/(short)0, c, /*reuse_a=*/false, /*reuse_b=*/false);
}

// A fragment (16x32 f16) from LDS, row-major rows of `rowstride` halves.
// contiguous K order used here (matches pack_b_kernel): lanes 0-15 -> M=lane, K = kbase+0..15;
//             lanes 16-31 -> M=lane-16, K = kbase+16..31.
__device__ inline v16h load_afrag(const _Float16* sA, int rowstride, int lane, int kbase) {
  int row = lane & 15;
  int kb  = lane >> 4;
  const _Float16* p = sA + row * rowstride + kbase + kb * 16;
  v8h lo = *(const v8h*)p;        // 16B-aligned LDS load
  v8h hi = *(const v8h*)(p + 8);  // second 8-half chunk
  v16h a;
#pragma unroll
  for (int i = 0; i < 8; i++) { a[i] = lo[i]; a[8 + i] = hi[i]; }
  return a;
}

// B fragment: weights pre-packed so each lane reads 16 contiguous halves (32B).
__device__ inline v16h load_bfrag(const _Float16* pk, int frag, int lane) {
  return *(const v16h*)(pk + (size_t)frag * 512 + (size_t)lane * 16);
}

__device__ inline float silu_f(float x) { return x / (1.0f + __expf(-x)); }

// ---------------- weight pre-packing ----------------

// Pack W[K][N] (f32, row-major) into WMMA-B fragment order (f16).
// Fragment (ks,nt): lanes 0-15 cover K = ks*32+0..15, lanes 16-31 cover K = ks*32+16..31,
// element i steps K, column = nt*16 + (lane&15).
__global__ void pack_b_kernel(const float* __restrict__ W, _Float16* __restrict__ out,
                              int K, int N) {
  int idx = blockIdx.x * blockDim.x + threadIdx.x;
  int total = (K / 32) * (N / 16) * 32;
  if (idx >= total) return;
  int lane = idx & 31;
  int frag = idx >> 5;
  int NT = N / 16;
  int ks = frag / NT, nt = frag - ks * NT;
  int kb = lane >> 4, col = lane & 15;
  _Float16* o = out + (size_t)frag * 512 + (size_t)lane * 16;
  const float* w = W + (size_t)(ks * 32 + kb * 16) * N + nt * 16 + col;
#pragma unroll
  for (int i = 0; i < 16; i++) o[i] = (_Float16)w[(size_t)i * N];
}

// Wfused[l] = W_rbf[64,256] @ We[l][256,256]  (tiny, plain VALU)
__global__ void fuse_kernel(const float* __restrict__ Wrbf, const float* __restrict__ We,
                            float* __restrict__ Wf) {
  int idx = blockIdx.x * blockDim.x + threadIdx.x;
  if (idx >= 3 * RAD * EDGE) return;
  int l = idx / (RAD * EDGE);
  int rem = idx - l * RAD * EDGE;
  int r = rem / EDGE, c = rem - r * EDGE;
  const float* we = We + (size_t)l * EDGE * EDGE;
  float s = 0.0f;
  for (int k = 0; k < EDGE; k++) s += Wrbf[r * EDGE + k] * we[(size_t)k * EDGE + c];
  Wf[idx] = s;
}

// ---------------- pooling atoms -> blocks ----------------

__global__ void pool_atoms(const float* __restrict__ H, const float* __restrict__ Z,
                           const int* __restrict__ bid, float* __restrict__ Hb,
                           float* __restrict__ Zb, float* __restrict__ cnt, int NA) {
  int g = blockIdx.x * blockDim.x + threadIdx.x;
  int a = g >> 5, lane = g & 31;
  if (a >= NA) return;
  int b = bid[a];
#pragma unroll
  for (int k = 0; k < 4; k++) {
    int c = lane + k * 32;
    atomicAdd(&Hb[(size_t)b * HID + c], H[(size_t)a * HID + c]);
  }
  if (lane < 3) atomicAdd(&Zb[(size_t)b * 3 + lane], Z[(size_t)a * 3 + lane]);
  if (lane == 0) atomicAdd(&cnt[b], 1.0f);
}

__global__ void finalize_blocks(float* __restrict__ Hb, float* __restrict__ Zb,
                                const float* __restrict__ cnt, float* __restrict__ h,
                                _Float16* __restrict__ h16, int NB) {
  int r = blockIdx.x, t = threadIdx.x; // block = 128 threads
  float c = fmaxf(cnt[r], 1.0f);
  size_t o = (size_t)r * HID + t;
  float v = Hb[o] / c;
  Hb[o] = v;        // Hb output (d_out segment 0)
  h[o]  = v;        // evolving h (f32)
  h16[o] = (_Float16)v;
  if (t < 3) Zb[(size_t)r * 3 + t] = Zb[(size_t)r * 3 + t] / c;
}

// ---------------- RBF edge features (f16) ----------------

__global__ void rbf_kernel(const float* __restrict__ Zb, const int* __restrict__ edges,
                           _Float16* __restrict__ rbf16, int E) {
  int g = blockIdx.x * blockDim.x + threadIdx.x;
  int e = g >> 5, lane = g & 31;
  if (e >= E) return;
  int s = edges[e], d = edges[E + e];
  float dx = Zb[(size_t)s * 3 + 0] - Zb[(size_t)d * 3 + 0];
  float dy = Zb[(size_t)s * 3 + 1] - Zb[(size_t)d * 3 + 1];
  float dz = Zb[(size_t)s * 3 + 2] - Zb[(size_t)d * 3 + 2];
  float dist = sqrtf(dx * dx + dy * dy + dz * dz + 1e-12f);
  const float step = CUTOFF_F / (float)(RAD - 1);
#pragma unroll
  for (int k = 0; k < 2; k++) {
    int rr = lane + k * 32;
    float diff = dist - (float)rr * step;
    rbf16[(size_t)e * RAD + rr] = (_Float16)__expf(-10.0f * diff * diff);
  }
}

// ---------------- message layer: m = silu(hs@Wm1 + hd@Wm2 + rbf@Wf); scatter-add ----------------
// 32 edges x 256 outputs per block. 8 waves; wave w owns N-tiles {2w, 2w+1} and both
// 16-row M-tiles, so each B fragment load feeds two WMMAs (2x weight reuse vs MTILE=16).

__global__ void __launch_bounds__(256)
msg_kernel(const _Float16* __restrict__ h16, const _Float16* __restrict__ rbf16,
           const _Float16* __restrict__ Wm1p, const _Float16* __restrict__ Wm2p,
           const _Float16* __restrict__ Wfp, const int* __restrict__ edges,
           float* __restrict__ agg, int E) {
  __shared__ _Float16 sSrc[MTILE * HID];
  __shared__ _Float16 sDst[MTILE * HID];
  __shared__ _Float16 sRbf[MTILE * RAD];
  __shared__ int sDstId[MTILE];

  int tid = threadIdx.x;
  int e0 = blockIdx.x * MTILE;
  int sub = tid & 15;

  // prefetch next tile's edge indices (lowers to global_prefetch_b8)
  if (tid == 0) {
    __builtin_prefetch(&edges[e0 + MTILE], 0, 1);
    __builtin_prefetch(&edges[E + e0 + MTILE], 0, 1);
  }

#pragma unroll
  for (int k = 0; k < 2; k++) {
    int er = (tid >> 4) + k * 16;       // 0..31
    int e = e0 + er;
    if (e < E) {
      int s = edges[e], d = edges[E + e];
      *(v8h*)&sSrc[er * HID + sub * 8] = *(const v8h*)&h16[(size_t)s * HID + sub * 8];
      *(v8h*)&sDst[er * HID + sub * 8] = *(const v8h*)&h16[(size_t)d * HID + sub * 8];
      if (sub < 8) *(v8h*)&sRbf[er * RAD + sub * 8] = *(const v8h*)&rbf16[(size_t)e * RAD + sub * 8];
      if (sub == 0) sDstId[er] = d;
    } else {
      v8h z = {};
      *(v8h*)&sSrc[er * HID + sub * 8] = z;
      *(v8h*)&sDst[er * HID + sub * 8] = z;
      if (sub < 8) *(v8h*)&sRbf[er * RAD + sub * 8] = z;
      if (sub == 0) sDstId[er] = -1;
    }
  }
  __syncthreads();

  int w = tid >> 5, lane = tid & 31;
  int nt0 = w * 2, nt1 = nt0 + 1;
  v8f acc00 = {}, acc01 = {};   // M-tile 0 x {nt0, nt1}
  v8f acc10 = {}, acc11 = {};   // M-tile 1 x {nt0, nt1}

#pragma unroll
  for (int ks = 0; ks < 4; ks++) {          // h[src] @ Wm1, K = 128
    v16h b0 = load_bfrag(Wm1p, ks * 16 + nt0, lane);
    v16h b1 = load_bfrag(Wm1p, ks * 16 + nt1, lane);
    v16h a0 = load_afrag(sSrc,            HID, lane, ks * 32);
    v16h a1 = load_afrag(sSrc + 16 * HID, HID, lane, ks * 32);
    acc00 = wmma16(a0, b0, acc00);
    acc01 = wmma16(a0, b1, acc01);
    acc10 = wmma16(a1, b0, acc10);
    acc11 = wmma16(a1, b1, acc11);
  }
#pragma unroll
  for (int ks = 0; ks < 4; ks++) {          // h[dst] @ Wm2, K = 128
    v16h b0 = load_bfrag(Wm2p, ks * 16 + nt0, lane);
    v16h b1 = load_bfrag(Wm2p, ks * 16 + nt1, lane);
    v16h a0 = load_afrag(sDst,            HID, lane, ks * 32);
    v16h a1 = load_afrag(sDst + 16 * HID, HID, lane, ks * 32);
    acc00 = wmma16(a0, b0, acc00);
    acc01 = wmma16(a0, b1, acc01);
    acc10 = wmma16(a1, b0, acc10);
    acc11 = wmma16(a1, b1, acc11);
  }
#pragma unroll
  for (int ks = 0; ks < 2; ks++) {          // rbf @ (W_rbf@We), K = 64
    v16h b0 = load_bfrag(Wfp, ks * 16 + nt0, lane);
    v16h b1 = load_bfrag(Wfp, ks * 16 + nt1, lane);
    v16h a0 = load_afrag(sRbf,            RAD, lane, ks * 32);
    v16h a1 = load_afrag(sRbf + 16 * RAD, RAD, lane, ks * 32);
    acc00 = wmma16(a0, b0, acc00);
    acc01 = wmma16(a0, b1, acc01);
    acc10 = wmma16(a1, b0, acc10);
    acc11 = wmma16(a1, b1, acc11);
  }

  int kb = lane >> 4, col = lane & 15;
#pragma unroll
  for (int v = 0; v < 8; v++) {             // M-tile 0 rows
    int row = v + kb * 8;                   // C/D layout: M = vgpr + 8*(lane>=16)
    int db = sDstId[row];
    if (db >= 0) {
      atomicAdd(&agg[(size_t)db * EDGE + nt0 * 16 + col], silu_f(acc00[v]));
      atomicAdd(&agg[(size_t)db * EDGE + nt1 * 16 + col], silu_f(acc01[v]));
    }
  }
#pragma unroll
  for (int v = 0; v < 8; v++) {             // M-tile 1 rows
    int row = 16 + v + kb * 8;
    int db = sDstId[row];
    if (db >= 0) {
      atomicAdd(&agg[(size_t)db * EDGE + nt0 * 16 + col], silu_f(acc10[v]));
      atomicAdd(&agg[(size_t)db * EDGE + nt1 * 16 + col], silu_f(acc11[v]));
    }
  }
}

// ---------------- node update: h += silu(agg @ Wupd) ----------------

__global__ void __launch_bounds__(256)
upd_kernel(const float* __restrict__ agg, const _Float16* __restrict__ Wupdp,
           float* __restrict__ h, _Float16* __restrict__ h16, int NB) {
  __shared__ _Float16 sA[16 * EDGE];
  int tid = threadIdx.x;
  int r0 = blockIdx.x * 16;
  int rr = tid >> 4, sub = tid & 15;
  int r = r0 + rr;
  _Float16* dp = &sA[rr * EDGE + sub * 16];
  if (r < NB) {
    const float* src = &agg[(size_t)r * EDGE + sub * 16];
#pragma unroll
    for (int i = 0; i < 16; i++) dp[i] = (_Float16)src[i];
  } else {
#pragma unroll
    for (int i = 0; i < 16; i++) dp[i] = (_Float16)0.0f;
  }
  __syncthreads();

  int w = tid >> 5, lane = tid & 31;        // 8 waves -> 8 N-tiles of 16 (N=128)
  v8f acc = {};
#pragma unroll
  for (int ks = 0; ks < 8; ks++) {          // K = 256
    v16h a = load_afrag(sA, EDGE, lane, ks * 32);
    acc = wmma16(a, load_bfrag(Wupdp, ks * 8 + w, lane), acc);
  }
  int kb = lane >> 4, col = lane & 15;
#pragma unroll
  for (int v = 0; v < 8; v++) {
    int r2 = r0 + v + kb * 8;
    if (r2 < NB) {
      size_t o = (size_t)r2 * HID + w * 16 + col;
      float nv = h[o] + silu_f(acc[v]);
      h[o] = nv;
      h16[o] = (_Float16)nv;
    }
  }
}

// ---------------- output projection: P = h @ W_out ----------------

__global__ void __launch_bounds__(256)
proj_kernel(const _Float16* __restrict__ h16, const _Float16* __restrict__ Woutp,
            float* __restrict__ P, int NB) {
  __shared__ _Float16 sA[16 * HID];
  int tid = threadIdx.x;
  int r0 = blockIdx.x * 16;
  int rr = tid >> 4, sub = tid & 15;
  int r = r0 + rr;
  if (r < NB) {
    *(v8h*)&sA[rr * HID + sub * 8] = *(const v8h*)&h16[(size_t)r * HID + sub * 8];
  } else {
    v8h z = {};
    *(v8h*)&sA[rr * HID + sub * 8] = z;
  }
  __syncthreads();

  int w = tid >> 5, lane = tid & 31;
  v8f acc = {};
#pragma unroll
  for (int ks = 0; ks < 4; ks++) {          // K = 128
    v16h a = load_afrag(sA, HID, lane, ks * 32);
    acc = wmma16(a, load_bfrag(Woutp, ks * 8 + w, lane), acc);
  }
  int kb = lane >> 4, col = lane & 15;
#pragma unroll
  for (int v = 0; v < 8; v++) {
    int r2 = r0 + v + kb * 8;
    if (r2 < NB) P[(size_t)r2 * HID + w * 16 + col] = acc[v];
  }
}

// ---------------- L2 normalize rows + scatter to graphs ----------------

__global__ void l2norm_scatter(const float* __restrict__ P, const int* __restrict__ batch_id,
                               float* __restrict__ outB, float* __restrict__ gsum, int NB) {
  int r = blockIdx.x, t = threadIdx.x;      // block = 128 threads
  float v = P[(size_t)r * HID + t];
  __shared__ float red[HID];
  red[t] = v * v;
  __syncthreads();
  for (int s = HID / 2; s > 0; s >>= 1) {
    if (t < s) red[t] += red[t + s];
    __syncthreads();
  }
  float inv = 1.0f / fmaxf(sqrtf(red[0]), 1e-12f);
  float b = v * inv;
  outB[(size_t)r * HID + t] = b;
  atomicAdd(&gsum[(size_t)batch_id[r] * HID + t], b);
}

__global__ void gnorm_kernel(const float* __restrict__ gsum, float* __restrict__ outG, int NG) {
  int r = blockIdx.x, t = threadIdx.x;
  float v = gsum[(size_t)r * HID + t];
  __shared__ float red[HID];
  red[t] = v * v;
  __syncthreads();
  for (int s = HID / 2; s > 0; s >>= 1) {
    if (t < s) red[t] += red[t + s];
    __syncthreads();
  }
  float inv = 1.0f / fmaxf(sqrtf(red[0]), 1e-12f);
  outG[(size_t)r * HID + t] = v * inv;
}

// ---------------- host launcher ----------------

extern "C" void kernel_launch(void* const* d_in, const int* in_sizes, int n_in,
                              void* d_out, int out_size, void* d_ws, size_t ws_size,
                              hipStream_t stream) {
  const float* H     = (const float*)d_in[0];
  const float* Z     = (const float*)d_in[1];
  const float* W_rbf = (const float*)d_in[2];
  const float* Wm1   = (const float*)d_in[3];
  const float* Wm2   = (const float*)d_in[4];
  const float* We    = (const float*)d_in[5];
  const float* Wupd  = (const float*)d_in[6];
  const float* Wout  = (const float*)d_in[7];
  const int* block_id = (const int*)d_in[8];
  const int* batch_id = (const int*)d_in[9];
  const int* edges    = (const int*)d_in[10];

  int NA = in_sizes[8];
  int NB = in_sizes[9];
  int E  = in_sizes[10] / 2;
  int NG = (out_size - 2 * NB * HID) / HID;

  float* outHb = (float*)d_out;
  float* outBR = outHb + (size_t)NB * HID;
  float* outGR = outBR + (size_t)NB * HID;

  char* ws = (char*)d_ws;
  size_t off = 0;
  auto alloc = [&](size_t bytes) -> void* {
    void* p = ws + off;
    off = (off + bytes + 255) & ~(size_t)255;
    return p;
  };
  float*    cnt   = (float*)alloc((size_t)NB * 4);
  float*    Zb    = (float*)alloc((size_t)NB * 3 * 4);
  float*    h     = (float*)alloc((size_t)NB * HID * 4);
  _Float16* h16   = (_Float16*)alloc((size_t)NB * HID * 2);
  float*    agg   = (float*)alloc((size_t)NB * EDGE * 4);   // reused as P later
  float*    gsum  = (float*)alloc((size_t)NG * HID * 4);
  _Float16* rbf16 = (_Float16*)alloc((size_t)E * RAD * 2);
  float*    Wf32  = (float*)alloc((size_t)3 * RAD * EDGE * 4);
  _Float16* Wm1p  = (_Float16*)alloc((size_t)3 * HID * EDGE * 2);
  _Float16* Wm2p  = (_Float16*)alloc((size_t)3 * HID * EDGE * 2);
  _Float16* Wfp   = (_Float16*)alloc((size_t)3 * RAD * EDGE * 2);
  _Float16* Wupdp = (_Float16*)alloc((size_t)3 * EDGE * HID * 2);
  _Float16* Woutp = (_Float16*)alloc((size_t)HID * HID * 2);
  (void)ws_size; (void)n_in;

  // zero accumulators (fresh each call; capture-safe stream memsets)
  hipMemsetAsync(outHb, 0, (size_t)NB * HID * 4, stream);
  hipMemsetAsync(cnt,   0, (size_t)NB * 4, stream);
  hipMemsetAsync(Zb,    0, (size_t)NB * 3 * 4, stream);
  hipMemsetAsync(gsum,  0, (size_t)NG * HID * 4, stream);

  // weight fusion + B-fragment packing (tiny)
  {
    int tot = (HID / 32) * (EDGE / 16) * 32;
    for (int l = 0; l < 3; l++) {
      pack_b_kernel<<<(tot + 255) / 256, 256, 0, stream>>>(
          Wm1 + (size_t)l * HID * EDGE, Wm1p + (size_t)l * HID * EDGE, HID, EDGE);
      pack_b_kernel<<<(tot + 255) / 256, 256, 0, stream>>>(
          Wm2 + (size_t)l * HID * EDGE, Wm2p + (size_t)l * HID * EDGE, HID, EDGE);
    }
    int totU = (EDGE / 32) * (HID / 16) * 32;
    for (int l = 0; l < 3; l++)
      pack_b_kernel<<<(totU + 255) / 256, 256, 0, stream>>>(
          Wupd + (size_t)l * EDGE * HID, Wupdp + (size_t)l * EDGE * HID, EDGE, HID);
    int totO = (HID / 32) * (HID / 16) * 32;
    pack_b_kernel<<<(totO + 255) / 256, 256, 0, stream>>>(Wout, Woutp, HID, HID);

    fuse_kernel<<<(3 * RAD * EDGE + 255) / 256, 256, 0, stream>>>(W_rbf, We, Wf32);
    int totF = (RAD / 32) * (EDGE / 16) * 32;
    for (int l = 0; l < 3; l++)
      pack_b_kernel<<<(totF + 255) / 256, 256, 0, stream>>>(
          Wf32 + (size_t)l * RAD * EDGE, Wfp + (size_t)l * RAD * EDGE, RAD, EDGE);
  }

  // pooling + edge features
  unsigned poolBlocks = (unsigned)(((size_t)NA * 32 + 255) / 256);
  pool_atoms<<<poolBlocks, 256, 0, stream>>>(H, Z, block_id, outHb, Zb, cnt, NA);
  finalize_blocks<<<NB, HID, 0, stream>>>(outHb, Zb, cnt, h, h16, NB);
  unsigned rbfBlocks = (unsigned)(((size_t)E * 32 + 255) / 256);
  rbf_kernel<<<rbfBlocks, 256, 0, stream>>>(Zb, edges, rbf16, E);

  // message passing layers
  int mblocks = (E + MTILE - 1) / MTILE;
  int ublocks = (NB + 15) / 16;
  for (int l = 0; l < 3; l++) {
    hipMemsetAsync(agg, 0, (size_t)NB * EDGE * 4, stream);
    msg_kernel<<<mblocks, 256, 0, stream>>>(
        h16, rbf16,
        Wm1p + (size_t)l * HID * EDGE, Wm2p + (size_t)l * HID * EDGE,
        Wfp + (size_t)l * RAD * EDGE, edges, agg, E);
    upd_kernel<<<ublocks, 256, 0, stream>>>(
        agg, Wupdp + (size_t)l * EDGE * HID, h, h16, NB);
  }

  // readout
  proj_kernel<<<ublocks, 256, 0, stream>>>(h16, Woutp, agg /*reuse as P*/, NB);
  l2norm_scatter<<<NB, HID, 0, stream>>>(agg, batch_id, outBR, gsum, NB);
  gnorm_kernel<<<NG, HID, 0, stream>>>(gsum, outGR, NG);
}
